// ChronoSynth_5995774345869
// MI455X (gfx1250) — matrix operand order repaired and checked by reading, not compiled, not run on hardware
//
#include <hip/hip_runtime.h>

// ---------------------------------------------------------------------------
// Types for CDNA5 WMMA (wave32, 16x16x32 bf16 -> f32)
// ---------------------------------------------------------------------------
typedef __attribute__((ext_vector_type(16))) __bf16          v16bf_t;
typedef __attribute__((ext_vector_type(16))) unsigned short  v16u_t;
typedef __attribute__((ext_vector_type(8)))  float           v8f_t;

__device__ __forceinline__ unsigned short f2bf(float f) {
  union { float f; unsigned int u; } v; v.f = f;
  unsigned int u = v.u;
  unsigned int r = u + 0x7FFFu + ((u >> 16) & 1u);   // RNE f32 -> bf16
  return (unsigned short)(r >> 16);
}
__device__ __forceinline__ float bf2f(unsigned short s) {
  union { unsigned int u; float f; } v; v.u = ((unsigned int)s) << 16; return v.f;
}
__device__ __forceinline__ float lrelu(float x) { return x >= 0.f ? x : 0.2f * x; }

// ---------------------------------------------------------------------------
// Shapes
// ---------------------------------------------------------------------------
#define BB    2
#define TT    4
#define NIMG  8            // B*T
#define CH32  32
#define HH    256
#define WW    256
#define HP    258          // padded 256+2
#define HO    512
#define WO    512
#define PLANE (HO*WO)      // 262144

// ---------------------------------------------------------------------------
// zero words
// ---------------------------------------------------------------------------
__global__ void k_zero(unsigned int* __restrict__ p, long long n) {
  long long i = (long long)blockIdx.x * 256 + threadIdx.x;
  if (i < n) p[i] = 0u;
}

// ---------------------------------------------------------------------------
// features (B,32,T,256,256) f32 -> padded channels-last bf16 (n=b*T+t,258,258,32)
// ---------------------------------------------------------------------------
__global__ void k_feat2buf(const float* __restrict__ F, unsigned short* __restrict__ out) {
  int id = blockIdx.x * 256 + threadIdx.x;       // NIMG*256*256
  int x = id & 255, y = (id >> 8) & 255, n = id >> 16;
  int b = n >> 2, t = n & 3;
  const float* src = F + (long long)(b * 128 + t) * 65536 + y * 256 + x;  // c-stride = 4*65536
  unsigned int pk[16];
#pragma unroll
  for (int c = 0; c < 32; c += 2) {
    unsigned short lo = f2bf(src[(long long)c * 262144]);
    unsigned short hi = f2bf(src[(long long)(c + 1) * 262144]);
    pk[c >> 1] = (unsigned int)lo | ((unsigned int)hi << 16);
  }
  unsigned short* dst = out + ((long long)n * (HP * HP) + (y + 1) * HP + (x + 1)) * 32;
  uint4* d4 = (uint4*)dst;
  d4[0] = make_uint4(pk[0], pk[1], pk[2], pk[3]);
  d4[1] = make_uint4(pk[4], pk[5], pk[6], pk[7]);
  d4[2] = make_uint4(pk[8], pk[9], pk[10], pk[11]);
  d4[3] = make_uint4(pk[12], pk[13], pk[14], pk[15]);
}

// ---------------------------------------------------------------------------
// fuse: 1x1 conv 132->32 + lrelu, write padded channels-last bf16 (B,258,258,32)
// ---------------------------------------------------------------------------
__global__ void k_fuse(const float* __restrict__ F, const float* __restrict__ Wf,
                       const float* __restrict__ Bf, const float* __restrict__ TS,
                       unsigned short* __restrict__ xb) {
  int id = blockIdx.x * 256 + threadIdx.x;       // B*256*256
  int x = id & 255, y = (id >> 8) & 255, b = id >> 16;
  float tt = TS[0];
  float tv[4] = { tt + 1.f, tt, 1.f - tt, 2.f - tt };
  float acc[32];
#pragma unroll
  for (int co = 0; co < 32; ++co) acc[co] = Bf[co];
#pragma unroll 1
  for (int t = 0; t < 4; ++t) {
    const float* src = F + (long long)(b * 128 + t) * 65536 + y * 256 + x;
#pragma unroll 1
    for (int c = 0; c < 32; ++c) {
      float fv = src[(long long)c * 262144];
      const float* wr = Wf + (t * 33 + c);
#pragma unroll
      for (int co = 0; co < 32; ++co) acc[co] += wr[co * 132] * fv;
    }
    float fv = tv[t];
    const float* wr = Wf + (t * 33 + 32);
#pragma unroll
    for (int co = 0; co < 32; ++co) acc[co] += wr[co * 132] * fv;
  }
  unsigned int pk[16];
#pragma unroll
  for (int c = 0; c < 32; c += 2)
    pk[c >> 1] = (unsigned int)f2bf(lrelu(acc[c])) | ((unsigned int)f2bf(lrelu(acc[c + 1])) << 16);
  unsigned short* dst = xb + ((long long)b * (HP * HP) + (y + 1) * HP + (x + 1)) * 32;
  uint4* d4 = (uint4*)dst;
  d4[0] = make_uint4(pk[0], pk[1], pk[2], pk[3]);
  d4[1] = make_uint4(pk[4], pk[5], pk[6], pk[7]);
  d4[2] = make_uint4(pk[8], pk[9], pk[10], pk[11]);
  d4[3] = make_uint4(pk[12], pk[13], pk[14], pk[15]);
}

// ---------------------------------------------------------------------------
// Repack f32 weights (Cout,32,3,3) into exact WMMA A-matrix order:
// AP[((tile*9+tap)*32+lane)*16+slot], ci = 16*(lane/16) + slot (contiguous K halves, matches B).
// ---------------------------------------------------------------------------
__global__ void k_repack(const float* __restrict__ Wsrc, unsigned short* __restrict__ AP,
                         int Cout, int tiles) {
  int id = blockIdx.x * 256 + threadIdx.x;
  int total = tiles * 9 * 32 * 16;
  if (id >= total) return;
  int s = id & 15;
  int lane = (id >> 4) & 31;
  int tap = (id >> 9) % 9;
  int tile = id / (9 * 512);
  int m = tile * 16 + (lane & 15);
  int base = (lane < 16) ? 0 : 16;
  int ci = base + s;
  int ky = tap / 3, kx = tap % 3;
  float v = (m < Cout) ? Wsrc[((m * 32 + ci) * 3 + ky) * 3 + kx] : 0.f;
  AP[id] = f2bf(v);
}

// ---------------------------------------------------------------------------
// 3x3 conv, Cin=32, bf16 in (padded channels-last) -> bf16 out, WMMA per tap.
// One wave = one 16-out-channel x 16-pixel tile; 9 x v_wmma_f32_16x16x32_bf16.
// ---------------------------------------------------------------------------
__global__ void __launch_bounds__(256)
k_conv3x3_wmma(const unsigned short* __restrict__ in,
               const unsigned short* __restrict__ AP,
               const float* __restrict__ bias,
               int Cout, int tiles, int do_relu,
               unsigned short* __restrict__ out,
               int out_row_stride, int out_base, long long out_img_stride) {
  const int lane = threadIdx.x & 31;
  const int wave = threadIdx.x >> 5;
  const int strip = blockIdx.x * 8 + wave;       // 16 strips of 16 px
  const int y = blockIdx.y;
  const int z = blockIdx.z;
  const int n = z / tiles;
  const int tile = z - n * tiles;
  const int col = lane & 15;
  const int kb = (lane < 16) ? 0 : 16;           // B-matrix K-half: contiguous k = kb + s, matches A repack
  const int x0 = strip * 16;
  const int mrow = tile * 16 + ((lane >= 16) ? 8 : 0);

  v8f_t acc;
#pragma unroll
  for (int r = 0; r < 8; ++r) {
    int m = mrow + r;
    acc[r] = (m < Cout) ? bias[m] : 0.f;
  }
  const unsigned short* ib = in + (long long)n * ((long long)HP * HP * 32)
                                + (long long)(x0 + col) * 32 + kb;
  const unsigned short* ap = AP + (long long)(tile * 9) * 512 + lane * 16;
#pragma unroll
  for (int tap = 0; tap < 9; ++tap) {
    const int ky = tap / 3, kx = tap % 3;
    v16u_t av = *(const v16u_t*)(ap + tap * 512);
    v16u_t bv = *(const v16u_t*)(ib + ((long long)(y + ky) * HP + kx) * 32);
    acc = __builtin_amdgcn_wmma_f32_16x16x32_bf16(
        false, __builtin_bit_cast(v16bf_t, av),
        false, __builtin_bit_cast(v16bf_t, bv),
        (short)0, acc, false, false);
  }
  if (do_relu) {
#pragma unroll
    for (int r = 0; r < 8; ++r) acc[r] = lrelu(acc[r]);
  }
  unsigned int pk[4];
#pragma unroll
  for (int r = 0; r < 8; r += 2)
    pk[r >> 1] = (unsigned int)f2bf(acc[r]) | ((unsigned int)f2bf(acc[r + 1]) << 16);
  unsigned short* op = out + (long long)n * out_img_stride + out_base
                     + (long long)y * out_row_stride + (long long)(x0 + col) * 32 + mrow;
  *(uint4*)op = make_uint4(pk[0], pk[1], pk[2], pk[3]);
}

// ---------------------------------------------------------------------------
// Transposed conv, stride 2, pad (1,2): in (n,256,256,ch-stride 32) bf16,
// out (n,512,512,C2) f32. Direct gather over valid parity taps.
// ---------------------------------------------------------------------------
template <int C2>
__global__ void __launch_bounds__(128)
k_convT(const unsigned short* __restrict__ in, const float* __restrict__ Wt,
        const float* __restrict__ Bt, float* __restrict__ out) {
  int X = blockIdx.x * 128 + threadIdx.x;
  int Y = blockIdx.y;
  int n = blockIdx.z;
  float acc[C2];
#pragma unroll
  for (int co = 0; co < C2; ++co) acc[co] = Bt[co];
  const unsigned short* ibase = in + (long long)n * ((long long)HH * WW * 32);
#pragma unroll
  for (int ky = 0; ky < 3; ++ky) {
    int d = Y - 1 + ky;
    if (d < 0 || (d & 1) || d >= 512) continue;
    int yy = d >> 1;
#pragma unroll
    for (int kx = 0; kx < 3; ++kx) {
      int e = X - 1 + kx;
      if (e < 0 || (e & 1) || e >= 512) continue;
      int xx = e >> 1;
      const unsigned short* ip = ibase + ((long long)yy * WW + xx) * 32;
      const float* wb = Wt + (ky * 3 + kx);
#pragma unroll 1
      for (int ci = 0; ci < C2; ++ci) {
        float iv = bf2f(ip[ci]);
        const float* wc = wb + ci * 9;
#pragma unroll
        for (int co = 0; co < C2; ++co) acc[co] += wc[co * C2 * 9] * iv;
      }
    }
  }
  float* op = out + ((long long)n * PLANE + (long long)Y * WO + X) * C2;
#pragma unroll
  for (int co = 0; co < C2; ++co) op[co] = acc[co];
}

// ---------------------------------------------------------------------------
// 3x3 conv at 512x512, f32 channels-last in/out, small channel counts (VALU)
// ---------------------------------------------------------------------------
template <int CIN, int COUT>
__global__ void __launch_bounds__(128)
k_conv3(const float* __restrict__ in, const float* __restrict__ Wc,
        const float* __restrict__ Bc, float* __restrict__ out) {
  int X = blockIdx.x * 128 + threadIdx.x;
  int Y = blockIdx.y;
  int n = blockIdx.z;
  float acc[COUT];
#pragma unroll
  for (int co = 0; co < COUT; ++co) acc[co] = Bc[co];
  const float* ibase = in + (long long)n * PLANE * CIN;
#pragma unroll
  for (int ky = 0; ky < 3; ++ky) {
    int yy = Y - 1 + ky;
    if (yy < 0 || yy >= 512) continue;
#pragma unroll
    for (int kx = 0; kx < 3; ++kx) {
      int xx = X - 1 + kx;
      if (xx < 0 || xx >= 512) continue;
      const float* ip = ibase + ((long long)yy * WO + xx) * CIN;
      const float* wb = Wc + (ky * 3 + kx);
#pragma unroll 1
      for (int ci = 0; ci < CIN; ++ci) {
        float iv = ip[ci];
#pragma unroll
        for (int co = 0; co < COUT; ++co) acc[co] += wb[(co * CIN + ci) * 9] * iv;
      }
    }
  }
  float* op = out + ((long long)n * PLANE + (long long)Y * WO + X) * COUT;
#pragma unroll
  for (int co = 0; co < COUT; ++co) op[co] = acc[co];
}

// ---------------------------------------------------------------------------
// per-pixel channel softmax, channels-last
// ---------------------------------------------------------------------------
template <int CH>
__global__ void k_softmax(float* __restrict__ p, long long npix) {
  long long id = (long long)blockIdx.x * 256 + threadIdx.x;
  if (id >= npix) return;
  float* q = p + id * CH;
  float v[CH];
  float mx = -3.4e38f;
#pragma unroll
  for (int c = 0; c < CH; ++c) { v[c] = q[c]; mx = fmaxf(mx, v[c]); }
  float s = 0.f;
#pragma unroll
  for (int c = 0; c < CH; ++c) { v[c] = expf(v[c] - mx); s += v[c]; }
  float r = 1.f / s;
#pragma unroll
  for (int c = 0; c < CH; ++c) q[c] = v[c] * r;
}

// ---------------------------------------------------------------------------
// Adaptive-kernel bilinear synthesis + occlusion blend over T frames
// frames (T,B,3,512,512); WG/AL/BE (N,512,512,9); OC (B,512,512,4)
// out (B,3,512,512)
// ---------------------------------------------------------------------------
__global__ void __launch_bounds__(128)
k_synth(const float* __restrict__ FR, const float* __restrict__ WG,
        const float* __restrict__ AL, const float* __restrict__ BE,
        const float* __restrict__ OC, float* __restrict__ out) {
  int X = blockIdx.x * 128 + threadIdx.x;
  int Y = blockIdx.y;
  int b = blockIdx.z;
  long long pix = (long long)Y * WO + X;
  float a0 = 0.f, a1 = 0.f, a2 = 0.f;
#pragma unroll 1
  for (int t = 0; t < 4; ++t) {
    float ov = OC[((long long)b * PLANE + pix) * 4 + t];
    long long pb = ((long long)(b * 4 + t) * PLANE + pix) * 9;
    const float* f0 = FR + (long long)((t * 2 + b) * 3) * PLANE;
    const float* f1 = f0 + PLANE;
    const float* f2 = f1 + PLANE;
#pragma unroll 1
    for (int k = 0; k < 9; ++k) {
      float wk = WG[pb + k] * ov;
      float al = AL[pb + k];
      float be = BE[pb + k];
      float py = (float)(Y + k / 3) + al;           // padded-frame coords
      float px = (float)(X + k % 3) + be;
      int y0 = (int)floorf(py); y0 = y0 < 0 ? 0 : (y0 > 512 ? 512 : y0);
      int x0 = (int)floorf(px); x0 = x0 < 0 ? 0 : (x0 > 512 ? 512 : x0);
      float wy = py - (float)y0; wy = wy < 0.f ? 0.f : (wy > 1.f ? 1.f : wy);
      float wx = px - (float)x0; wx = wx < 0.f ? 0.f : (wx > 1.f ? 1.f : wx);
      int r0 = y0 - 1; r0 = r0 < 0 ? 0 : (r0 > 511 ? 511 : r0);
      int r1 = y0 > 511 ? 511 : y0;                 // clamp of (y0+1)-1
      int c0 = x0 - 1; c0 = c0 < 0 ? 0 : (c0 > 511 ? 511 : c0);
      int c1 = x0 > 511 ? 511 : x0;
      float w00 = (1.f - wy) * (1.f - wx), w01 = (1.f - wy) * wx;
      float w10 = wy * (1.f - wx),         w11 = wy * wx;
      long long i00 = (long long)r0 * WO + c0, i01 = (long long)r0 * WO + c1;
      long long i10 = (long long)r1 * WO + c0, i11 = (long long)r1 * WO + c1;
      a0 += wk * (f0[i00] * w00 + f0[i01] * w01 + f0[i10] * w10 + f0[i11] * w11);
      a1 += wk * (f1[i00] * w00 + f1[i01] * w01 + f1[i10] * w10 + f1[i11] * w11);
      a2 += wk * (f2[i00] * w00 + f2[i01] * w01 + f2[i10] * w10 + f2[i11] * w11);
    }
  }
  long long ob = (long long)(b * 3) * PLANE + pix;
  out[ob] = a0;
  out[ob + PLANE] = a1;
  out[ob + 2 * PLANE] = a2;
}

// ---------------------------------------------------------------------------
// Launcher
// ---------------------------------------------------------------------------
extern "C" void kernel_launch(void* const* d_in, const int* in_sizes, int n_in,
                              void* d_out, int out_size, void* d_ws, size_t ws_size,
                              hipStream_t stream) {
  (void)in_sizes; (void)n_in; (void)out_size; (void)ws_size;

  const float* F  = (const float*)d_in[0];   // features
  const float* FR = (const float*)d_in[1];   // frames
  const float* TS = (const float*)d_in[2];   // time scalar
  const float* FW = (const float*)d_in[3];   // fuse_w
  const float* FBs = (const float*)d_in[4];  // fuse_b
  const float* P[3][8];
  for (int s = 0; s < 3; ++s)
    for (int j = 0; j < 8; ++j) P[s][j] = (const float*)d_in[5 + s * 8 + j];
  const float* O[8];
  for (int j = 0; j < 8; ++j) O[j] = (const float*)d_in[29 + j];

  // workspace layout (256B aligned; sizes are already multiples of 256)
  char* ws = (char*)d_ws;
  size_t cur = 0;
  auto take = [&](size_t bytes) -> char* {
    char* r = ws + cur;
    cur += (bytes + 255) & ~(size_t)255;
    return r;
  };
  const size_t PADIMG = (size_t)HP * HP * 32 * 2;        // padded bf16 image
  unsigned short* XB = (unsigned short*)take((size_t)BB * PADIMG);
  unsigned short* FB = (unsigned short*)take((size_t)NIMG * PADIMG);
  unsigned short* T1 = (unsigned short*)take((size_t)NIMG * PADIMG);
  unsigned short* T2 = (unsigned short*)take((size_t)NIMG * HH * WW * 32 * 2);
  float* T3 = (float*)take((size_t)NIMG * PLANE * 9 * 4);  // >= 2*PLANE*32*4
  float* WGp = (float*)take((size_t)NIMG * PLANE * 9 * 4);
  float* ALp = (float*)take((size_t)NIMG * PLANE * 9 * 4);
  float* BEp = (float*)take((size_t)NIMG * PLANE * 9 * 4);
  float* OCp = (float*)take((size_t)BB * PLANE * 4 * 4);
  unsigned short* APk = (unsigned short*)take(32768);

  // zero padded buffers (XB,FB,T1 contiguous) so conv zero-padding borders hold
  long long zwords = (long long)((BB + NIMG + NIMG) * PADIMG) / 4;
  k_zero<<<(unsigned)((zwords + 255) / 256), 256, 0, stream>>>((unsigned int*)XB, zwords);

  k_feat2buf<<<NIMG * 256, 256, 0, stream>>>(F, FB);
  k_fuse<<<BB * 256, 256, 0, stream>>>(F, FW, FBs, TS, XB);

  auto conv_wmma = [&](const unsigned short* in, const float* wsrc, const float* bsrc,
                       int Cout, int nimg, unsigned short* out, int padded_out) {
    int tiles = (Cout + 15) / 16;
    int total = tiles * 9 * 32 * 16;
    k_repack<<<(total + 255) / 256, 256, 0, stream>>>(wsrc, APk, Cout, tiles);
    long long ois = padded_out ? (long long)HP * HP * 32 : (long long)HH * WW * 32;
    int ors = padded_out ? HP * 32 : WW * 32;
    int obase = padded_out ? (HP + 1) * 32 : 0;
    dim3 g(2, 256, nimg * tiles);
    k_conv3x3_wmma<<<g, 256, 0, stream>>>(in, APk, bsrc, Cout, tiles, 1,
                                          out, ors, obase, ois);
  };

  // occlusion subnet (2 images from fused x)
  conv_wmma(XB, O[0], O[1], 32, BB, T1, 1);
  conv_wmma(T1, O[2], O[3], 32, BB, T2, 0);
  k_convT<32><<<dim3(4, 512, BB), 128, 0, stream>>>(T2, O[4], O[5], T3);
  k_conv3<32, 4><<<dim3(4, 512, BB), 128, 0, stream>>>(T3, O[6], O[7], OCp);
  k_softmax<4><<<(unsigned)(((long long)BB * PLANE + 255) / 256), 256, 0, stream>>>(
      OCp, (long long)BB * PLANE);

  // weight / alpha / beta subnets (8 images from features)
  float* outs[3] = { WGp, ALp, BEp };
  for (int s = 0; s < 3; ++s) {
    conv_wmma(FB, P[s][0], P[s][1], 32, NIMG, T1, 1);
    conv_wmma(T1, P[s][2], P[s][3], 9, NIMG, T2, 0);   // ch 9..15 garbage, never read
    k_convT<9><<<dim3(4, 512, NIMG), 128, 0, stream>>>(T2, P[s][4], P[s][5], T3);
    k_conv3<9, 9><<<dim3(4, 512, NIMG), 128, 0, stream>>>(T3, P[s][6], P[s][7], outs[s]);
  }
  k_softmax<9><<<(unsigned)(((long long)NIMG * PLANE + 255) / 256), 256, 0, stream>>>(
      WGp, (long long)NIMG * PLANE);

  // final synthesis + occlusion blend
  k_synth<<<dim3(4, 512, BB), 128, 0, stream>>>(FR, WGp, ALp, BEp, OCp, (float*)d_out);
}
